// GNNLayer_10462540333147
// MI455X (gfx1250) — matrix operand order repaired and checked by reading, not compiled, not run on hardware
//
#include <hip/hip_runtime.h>
#include <stdint.h>

// ---------------------------------------------------------------------------
// Types for CDNA5 WMMA (gfx1250, wave32)
// ---------------------------------------------------------------------------
typedef __attribute__((ext_vector_type(16))) __bf16 v16bf;
typedef __attribute__((ext_vector_type(8)))  float  v8f;

#define LDA     296      // bf16 elements per LDS activation row (bank-spread stride)
#define TILE_M  64       // rows (edges/nodes) per block
#define WAVES   8
#define THREADS 256

// Model dims (fixed by reference)
#define D_NODE 128
#define D_EDGE 16
#define MSG_DIM 128
#define OUT_DIM 128

static __device__ __forceinline__ unsigned short f32_to_bf16(float f) {
    unsigned u = __float_as_uint(f);
    u += 0x7FFFu + ((u >> 16) & 1u);     // round-to-nearest-even
    return (unsigned short)(u >> 16);
}
static __device__ __forceinline__ float bf16_to_f32(unsigned short h) {
    return __uint_as_float(((unsigned)h) << 16);
}
static __device__ __forceinline__ unsigned pack_bf16x2(float a, float b) {
    return (unsigned)f32_to_bf16(a) | ((unsigned)f32_to_bf16(b) << 16);
}

// Async 16B global -> LDS copy (CDNA5 ASYNCcnt path, no VGPR round trip)
static __device__ __forceinline__ void async_g2l_b128(unsigned lds_off, const void* gsrc) {
    asm volatile("global_load_async_to_lds_b128 %0, %1, off"
                 :: "v"(lds_off), "v"((unsigned long long)(size_t)gsrc)
                 : "memory");
}
static __device__ __forceinline__ void wait_asynccnt0() {
    asm volatile("s_wait_asynccnt 0x0" ::: "memory");
}

// A-fragment (16x32 bf16) from LDS tile, ISA layout:
// lane l: row = l&15, h = l>>4; VGPR0-3 = K{8h..8h+7}, VGPR4-7 = K{16+8h..16+8h+7}
static __device__ __forceinline__ v16bf load_a_frag(const unsigned short* bufA,
                                                    int mt, int kt, int lane) {
    int row = mt * 16 + (lane & 15);
    int h   = lane >> 4;
    const unsigned short* p = bufA + row * LDA + kt * 32;
    union { uint4 u[2]; v16bf v; } r;
    r.u[0] = *(const uint4*)(p + 8 * h);        // K = 8h .. 8h+7
    r.u[1] = *(const uint4*)(p + 16 + 8 * h);   // K = 16+8h .. 16+8h+7
    return r.v;
}

// B-fragment (32x16 bf16) from global transposed weights Wt[n][k] (stride Kp):
// lane l: col = l&15, h = l>>4; elements 0-7 = K{8h..8h+7}, 8-15 = K{16+8h..16+8h+7}.
static __device__ __forceinline__ v16bf load_b_frag(const unsigned short* __restrict__ Wt,
                                                    int Kp, int nt, int kt, int lane) {
    int n = lane & 15;
    int h = lane >> 4;
    const unsigned short* p = Wt + (size_t)(nt * 16 + n) * Kp + kt * 32;
    union { uint4 u[2]; v16bf v; } r;
    r.u[0] = *(const uint4*)(p + 8 * h);
    r.u[1] = *(const uint4*)(p + 16 + 8 * h);
    return r.v;
}

// One MLP layer over the 64-row LDS activation tile. 8 waves; each wave owns
// row-tiles mt=0..3 and col-tiles nt = wave + 8*j (j < NT/8). K = KT*32.
// kt-outer loop: A-frags loaded once per (mt,kt), B-frags once per (nt,kt).
template<int NT, int KT, bool RELU, bool TO_LDS, bool TO_GBF, bool TO_GF32>
static __device__ __forceinline__ void mlp_layer(
    unsigned short* bufA, const unsigned short* __restrict__ Wt, int Kp,
    const float* __restrict__ bias,
    unsigned short* __restrict__ gbf, float* __restrict__ gf32, int gStride,
    long long rowBase, long long rowLimit,
    int lane, int wave)
{
    constexpr int ND = NT / 8;       // distinct n-tiles per wave
    v8f acc[4][ND];
#pragma unroll
    for (int mt = 0; mt < 4; ++mt)
#pragma unroll
        for (int j = 0; j < ND; ++j)
#pragma unroll
            for (int q = 0; q < 8; ++q) acc[mt][j][q] = 0.0f;

#pragma unroll
    for (int kt = 0; kt < KT; ++kt) {
        v16bf a[4];
#pragma unroll
        for (int mt = 0; mt < 4; ++mt) a[mt] = load_a_frag(bufA, mt, kt, lane);
        v16bf b[ND];
#pragma unroll
        for (int j = 0; j < ND; ++j) b[j] = load_b_frag(Wt, Kp, wave + 8 * j, kt, lane);
#pragma unroll
        for (int mt = 0; mt < 4; ++mt)
#pragma unroll
            for (int j = 0; j < ND; ++j)
                acc[mt][j] = __builtin_amdgcn_wmma_f32_16x16x32_bf16(
                    false, a[mt], false, b[j], (short)0, acc[mt][j], false, false);
    }

    __syncthreads();   // all reads of bufA done before overwrite

    int n = lane & 15;
    int h = lane >> 4;
#pragma unroll
    for (int mt = 0; mt < 4; ++mt) {
#pragma unroll
        for (int j = 0; j < ND; ++j) {
            int nt  = wave + 8 * j;
            int col = nt * 16 + n;
            float bv = bias[col];
#pragma unroll
            for (int q = 0; q < 8; ++q) {
                int row = mt * 16 + 8 * h + q;   // C/D layout: lane holds col n, rows 8h+q
                float v = acc[mt][j][q] + bv;
                if (RELU) v = v > 0.0f ? v : 0.0f;
                if (TO_LDS) bufA[row * LDA + col] = f32_to_bf16(v);
                long long gr = rowBase + row;
                if (TO_GBF)  { if (gr < rowLimit) gbf [gr * (long long)gStride + col] = f32_to_bf16(v); }
                if (TO_GF32) { if (gr < rowLimit) gf32[gr * (long long)gStride + col] = v; }
            }
        }
    }
    __syncthreads();
}

// ---------------------------------------------------------------------------
// Kernel 0: conversions
// ---------------------------------------------------------------------------
__global__ void cvt_f32_bf16_kernel(const float* __restrict__ src,
                                    unsigned short* __restrict__ dst, long long n) {
    long long i = (long long)blockIdx.x * blockDim.x + threadIdx.x;
    if (i < n) dst[i] = f32_to_bf16(src[i]);
}

// src is [K][Nn] row-major (fin x fout); dst is Wt[n][k] with k padded to Kp (zeros).
__global__ void wt_transpose_kernel(const float* __restrict__ src,
                                    unsigned short* __restrict__ dst,
                                    int K, int Nn, int Kp) {
    long long idx = (long long)blockIdx.x * blockDim.x + threadIdx.x;
    long long total = (long long)Nn * Kp;
    if (idx >= total) return;
    int n = (int)(idx / Kp);
    int k = (int)(idx - (long long)n * Kp);
    dst[idx] = (k < K) ? f32_to_bf16(src[(size_t)k * Nn + n]) : (unsigned short)0;
}

__global__ void aggr_init_kernel(float* __restrict__ segmax, float* __restrict__ denom,
                                 float* __restrict__ aggr, long long N) {
    long long i = (long long)blockIdx.x * blockDim.x + threadIdx.x;
    if (i < N * D_NODE) aggr[i] = 0.0f;
    if (i < N) { segmax[i] = -3.0e38f; denom[i] = 0.0f; }
}

// ---------------------------------------------------------------------------
// Kernel 1: fused edge message MLP (272->256->256->128) + gate MLP (128->128->128->1)
// ---------------------------------------------------------------------------
__global__ void __launch_bounds__(THREADS)
edge_mlp_kernel(const unsigned short* __restrict__ nodes_bf,
                const unsigned short* __restrict__ edges_bf,
                const int* __restrict__ senders,
                const int* __restrict__ receivers,
                const unsigned short* __restrict__ W1t, const float* __restrict__ b1,
                const unsigned short* __restrict__ W2t, const float* __restrict__ b2,
                const unsigned short* __restrict__ W3t, const float* __restrict__ b3,
                const unsigned short* __restrict__ A1t, const float* __restrict__ ab1,
                const unsigned short* __restrict__ A2t, const float* __restrict__ ab2,
                const float* __restrict__ ag_w3, const float* __restrict__ ag_b3,
                unsigned short* __restrict__ msgs_bf,
                float* __restrict__ gate,
                long long E)
{
    __shared__ alignas(16) unsigned short bufA[TILE_M * LDA];
    __shared__ int   s_snd[TILE_M];
    __shared__ int   s_rcv[TILE_M];
    __shared__ float s_agw3[128];

    const int tid  = threadIdx.x;
    const int lane = tid & 31;
    const int wave = tid >> 5;
    const long long base = (long long)blockIdx.x * TILE_M;

    if (tid < TILE_M) {
        long long e = base + tid;
        s_snd[tid] = (e < E) ? senders[e]   : 0;
        s_rcv[tid] = (e < E) ? receivers[e] : 0;
    }
    if (tid < 128) s_agw3[tid] = ag_w3[tid];
    __syncthreads();

    // Build feats tile [64][288] bf16 = concat(edge16, sender128, receiver128, pad16)
    // as 36 x 16B chunks per row: async global->LDS for real data, zero-store for pad.
    const unsigned ldsBase = (unsigned)(size_t)(&bufA[0]);
    for (int idx = tid; idx < TILE_M * 36; idx += THREADS) {
        int row = idx / 36;
        int c   = idx - row * 36;
        bool ok = (base + row) < E;
        unsigned lds = ldsBase + (unsigned)(row * (LDA * 2) + c * 16);
        if (c < 34 && ok) {
            const unsigned short* src;
            if (c < 2)       src = edges_bf + (size_t)(base + row) * D_EDGE + c * 8;
            else if (c < 18) src = nodes_bf + (size_t)s_snd[row] * D_NODE + (c - 2) * 8;
            else             src = nodes_bf + (size_t)s_rcv[row] * D_NODE + (c - 18) * 8;
            async_g2l_b128(lds, src);
        } else {
            uint4 z = {0u, 0u, 0u, 0u};
            *(uint4*)(bufA + row * LDA + c * 8) = z;
        }
    }
    wait_asynccnt0();
    __syncthreads();

    // message MLP
    mlp_layer<16, 9, true,  true,  false, false>(bufA, W1t, 288, b1, nullptr, nullptr, 0, base, E, lane, wave);
    mlp_layer<16, 8, false, true,  false, false>(bufA, W2t, 256, b2, nullptr, nullptr, 0, base, E, lane, wave);
    mlp_layer< 8, 8, false, true,  true,  false>(bufA, W3t, 256, b3, msgs_bf, nullptr, MSG_DIM, base, E, lane, wave);
    // gate MLP on msgs (still in bufA cols 0..127)
    mlp_layer< 8, 4, true,  true,  false, false>(bufA, A1t, 128, ab1, nullptr, nullptr, 0, base, E, lane, wave);
    mlp_layer< 8, 4, false, true,  false, false>(bufA, A2t, 128, ab2, nullptr, nullptr, 0, base, E, lane, wave);

    // gate logit: dot(g2[e], ag_w3) + ag_b3   (vectorized LDS reads)
    if (tid < TILE_M) {
        long long e = base + tid;
        if (e < E) {
            float s = 0.0f;
#pragma unroll
            for (int c = 0; c < 16; ++c) {
                uint4 q = *(const uint4*)(bufA + tid * LDA + c * 8);
                const unsigned w[4] = {q.x, q.y, q.z, q.w};
#pragma unroll
                for (int t = 0; t < 4; ++t) {
                    s += bf16_to_f32((unsigned short)(w[t] & 0xFFFFu)) * s_agw3[c * 8 + 2 * t];
                    s += bf16_to_f32((unsigned short)(w[t] >> 16))     * s_agw3[c * 8 + 2 * t + 1];
                }
            }
            gate[e] = s + ag_b3[0];
        }
    }
}

// ---------------------------------------------------------------------------
// Kernel 2: segment softmax + weighted aggregation
// ---------------------------------------------------------------------------
static __device__ __forceinline__ void atomicMaxF(float* addr, float v) {
    if (v >= 0.0f) atomicMax((int*)addr, __float_as_int(v));
    else           atomicMin((unsigned int*)addr, __float_as_uint(v));
}

__global__ void seg_max_kernel(const float* __restrict__ gate,
                               const int* __restrict__ recv,
                               float* __restrict__ segmax, long long E) {
    long long e = (long long)blockIdx.x * blockDim.x + threadIdx.x;
    if (e < E) atomicMaxF(&segmax[recv[e]], gate[e]);
}

__global__ void exp_w_kernel(const float* __restrict__ gate,
                             const int* __restrict__ recv,
                             const float* __restrict__ segmax,
                             float* __restrict__ wexp,
                             float* __restrict__ denom, long long E) {
    long long e = (long long)blockIdx.x * blockDim.x + threadIdx.x;
    if (e < E) {
        int r = recv[e];
        float wv = __expf(gate[e] - segmax[r]);
        wexp[e] = wv;
        atomicAdd(&denom[r], wv);
    }
}

__global__ void scatter_kernel(const unsigned short* __restrict__ msgs_bf,
                               const int* __restrict__ recv,
                               const float* __restrict__ wexp,
                               const float* __restrict__ denom,
                               float* __restrict__ aggr, long long E) {
    long long idx = (long long)blockIdx.x * blockDim.x + threadIdx.x;
    if (idx < E * MSG_DIM) {
        long long e = idx >> 7;
        int d = (int)(idx & 127);
        int r = recv[e];
        float a = wexp[e] / denom[r];
        atomicAdd(&aggr[(size_t)r * MSG_DIM + d], a * bf16_to_f32(msgs_bf[idx]));
    }
}

// ---------------------------------------------------------------------------
// Kernel 3: node update MLP (256->256->256->128) -> d_out (f32)
// ---------------------------------------------------------------------------
__global__ void __launch_bounds__(THREADS)
node_mlp_kernel(const unsigned short* __restrict__ nodes_bf,
                const float* __restrict__ aggr,
                const unsigned short* __restrict__ U1t, const float* __restrict__ ub1,
                const unsigned short* __restrict__ U2t, const float* __restrict__ ub2,
                const unsigned short* __restrict__ U3t, const float* __restrict__ ub3,
                float* __restrict__ out, long long N)
{
    __shared__ alignas(16) unsigned short bufA[TILE_M * LDA];

    const int tid  = threadIdx.x;
    const int lane = tid & 31;
    const int wave = tid >> 5;
    const long long base = (long long)blockIdx.x * TILE_M;
    const unsigned ldsBase = (unsigned)(size_t)(&bufA[0]);

    // cols 0..127: node features, bf16 already -> async global->LDS (16 chunks/row)
    for (int idx = tid; idx < TILE_M * 16; idx += THREADS) {
        int row = idx >> 4;
        int c   = idx & 15;
        long long nd = base + row;
        unsigned lds = ldsBase + (unsigned)(row * (LDA * 2) + c * 16);
        if (nd < N) {
            async_g2l_b128(lds, nodes_bf + nd * D_NODE + c * 8);
        } else {
            uint4 z = {0u, 0u, 0u, 0u};
            *(uint4*)(bufA + row * LDA + c * 8) = z;
        }
    }
    // cols 128..255: aggr (f32 -> bf16 convert + pack, 16 chunks/row)
    for (int idx = tid; idx < TILE_M * 16; idx += THREADS) {
        int row = idx >> 4;
        int c   = idx & 15;
        long long nd = base + row;
        uint4 q = {0u, 0u, 0u, 0u};
        if (nd < N) {
            float4 f0 = *(const float4*)(aggr + nd * MSG_DIM + c * 8);
            float4 f1 = *(const float4*)(aggr + nd * MSG_DIM + c * 8 + 4);
            q.x = pack_bf16x2(f0.x, f0.y);
            q.y = pack_bf16x2(f0.z, f0.w);
            q.z = pack_bf16x2(f1.x, f1.y);
            q.w = pack_bf16x2(f1.z, f1.w);
        }
        *(uint4*)(bufA + row * LDA + 128 + c * 8) = q;
    }
    wait_asynccnt0();
    __syncthreads();

    mlp_layer<16, 8, true,  true,  false, false>(bufA, U1t, 256, ub1, nullptr, nullptr, 0, base, N, lane, wave);
    mlp_layer<16, 8, false, true,  false, false>(bufA, U2t, 256, ub2, nullptr, nullptr, 0, base, N, lane, wave);
    mlp_layer< 8, 8, false, false, false, true >(bufA, U3t, 256, ub3, nullptr, out, OUT_DIM, base, N, lane, wave);
}

// ---------------------------------------------------------------------------
// Host driver
// ---------------------------------------------------------------------------
extern "C" void kernel_launch(void* const* d_in, const int* in_sizes, int n_in,
                              void* d_out, int out_size, void* d_ws, size_t ws_size,
                              hipStream_t stream) {
    const float* nodes   = (const float*)d_in[0];
    const float* edges   = (const float*)d_in[1];
    const int*   senders = (const int*)d_in[2];
    const int*   recv    = (const int*)d_in[3];
    const float* msg_w1 = (const float*)d_in[4];  const float* msg_b1 = (const float*)d_in[5];
    const float* msg_w2 = (const float*)d_in[6];  const float* msg_b2 = (const float*)d_in[7];
    const float* msg_w3 = (const float*)d_in[8];  const float* msg_b3 = (const float*)d_in[9];
    const float* ag_w1  = (const float*)d_in[10]; const float* ag_b1  = (const float*)d_in[11];
    const float* ag_w2  = (const float*)d_in[12]; const float* ag_b2  = (const float*)d_in[13];
    const float* ag_w3  = (const float*)d_in[14]; const float* ag_b3  = (const float*)d_in[15];
    const float* up_w1  = (const float*)d_in[16]; const float* up_b1  = (const float*)d_in[17];
    const float* up_w2  = (const float*)d_in[18]; const float* up_b2  = (const float*)d_in[19];
    const float* up_w3  = (const float*)d_in[20]; const float* up_b3  = (const float*)d_in[21];

    const long long N = in_sizes[0] / D_NODE;
    const long long E = in_sizes[2];

    // workspace carve-out (256B aligned slabs)
    char* p = (char*)d_ws;
    auto carve = [&](size_t bytes) -> void* {
        void* r = (void*)p;
        p += (bytes + 255) & ~(size_t)255;
        return r;
    };
    unsigned short* nodes_bf = (unsigned short*)carve((size_t)N * D_NODE * 2);
    unsigned short* edges_bf = (unsigned short*)carve((size_t)E * D_EDGE * 2);
    unsigned short* W1t = (unsigned short*)carve((size_t)256 * 288 * 2);
    unsigned short* W2t = (unsigned short*)carve((size_t)256 * 256 * 2);
    unsigned short* W3t = (unsigned short*)carve((size_t)128 * 256 * 2);
    unsigned short* A1t = (unsigned short*)carve((size_t)128 * 128 * 2);
    unsigned short* A2t = (unsigned short*)carve((size_t)128 * 128 * 2);
    unsigned short* U1t = (unsigned short*)carve((size_t)256 * 256 * 2);
    unsigned short* U2t = (unsigned short*)carve((size_t)256 * 256 * 2);
    unsigned short* U3t = (unsigned short*)carve((size_t)128 * 256 * 2);
    unsigned short* msgs_bf = (unsigned short*)carve((size_t)E * MSG_DIM * 2);
    float* gate   = (float*)carve((size_t)E * 4);
    float* wexp   = (float*)carve((size_t)E * 4);
    float* segmax = (float*)carve((size_t)N * 4);
    float* denom  = (float*)carve((size_t)N * 4);
    float* aggr   = (float*)carve((size_t)N * MSG_DIM * 4);
    (void)ws_size; (void)n_in; (void)out_size;

    const int T = 256;
    // 0) conversions
    cvt_f32_bf16_kernel<<<(unsigned)((N * D_NODE + T - 1) / T), T, 0, stream>>>(nodes, nodes_bf, N * D_NODE);
    cvt_f32_bf16_kernel<<<(unsigned)((E * D_EDGE + T - 1) / T), T, 0, stream>>>(edges, edges_bf, E * D_EDGE);
    wt_transpose_kernel<<<(256 * 288 + T - 1) / T, T, 0, stream>>>(msg_w1, W1t, 272, 256, 288);
    wt_transpose_kernel<<<(256 * 256 + T - 1) / T, T, 0, stream>>>(msg_w2, W2t, 256, 256, 256);
    wt_transpose_kernel<<<(128 * 256 + T - 1) / T, T, 0, stream>>>(msg_w3, W3t, 256, 128, 256);
    wt_transpose_kernel<<<(128 * 128 + T - 1) / T, T, 0, stream>>>(ag_w1, A1t, 128, 128, 128);
    wt_transpose_kernel<<<(128 * 128 + T - 1) / T, T, 0, stream>>>(ag_w2, A2t, 128, 128, 128);
    wt_transpose_kernel<<<(256 * 256 + T - 1) / T, T, 0, stream>>>(up_w1, U1t, 256, 256, 256);
    wt_transpose_kernel<<<(256 * 256 + T - 1) / T, T, 0, stream>>>(up_w2, U2t, 256, 256, 256);
    wt_transpose_kernel<<<(128 * 256 + T - 1) / T, T, 0, stream>>>(up_w3, U3t, 256, 128, 256);
    aggr_init_kernel<<<(unsigned)((N * D_NODE + T - 1) / T), T, 0, stream>>>(segmax, denom, aggr, N);

    // 1) fused edge message + gate MLP (WMMA)
    edge_mlp_kernel<<<(unsigned)((E + TILE_M - 1) / TILE_M), THREADS, 0, stream>>>(
        nodes_bf, edges_bf, senders, recv,
        W1t, msg_b1, W2t, msg_b2, W3t, msg_b3,
        A1t, ag_b1, A2t, ag_b2, ag_w3, ag_b3,
        msgs_bf, gate, E);

    // 2) segment softmax + aggregation
    seg_max_kernel<<<(unsigned)((E + T - 1) / T), T, 0, stream>>>(gate, recv, segmax, E);
    exp_w_kernel  <<<(unsigned)((E + T - 1) / T), T, 0, stream>>>(gate, recv, segmax, wexp, denom, E);
    scatter_kernel<<<(unsigned)((E * MSG_DIM + T - 1) / T), T, 0, stream>>>(msgs_bf, recv, wexp, denom, aggr, E);

    // 3) node update MLP (WMMA) -> d_out
    node_mlp_kernel<<<(unsigned)((N + TILE_M - 1) / TILE_M), THREADS, 0, stream>>>(
        nodes_bf, aggr, U1t, up_b1, U2t, up_b2, U3t, up_b3, (float*)d_out, N);
}
